// S2HVQ_73495480369758
// MI455X (gfx1250) — matrix operand order repaired and checked by reading, not compiled, not run on hardware
//
#include <hip/hip_runtime.h>
#include <hip/hip_bf16.h>

typedef __attribute__((ext_vector_type(16))) _Float16 v16h;
typedef __attribute__((ext_vector_type(8)))  float    v8f;

static __device__ __forceinline__ v8f wmma_f16(v16h a, v16h b, v8f c) {
    // D = A(16x32 f16) * B(32x16 f16) + C(16x16 f32)
    return __builtin_amdgcn_wmma_f32_16x16x32_f16(
        /*neg_a=*/false, a, /*neg_b=*/false, b,
        /*c_mod=*/(short)0, c, /*reuse_a=*/false, /*reuse_b=*/false);
}

// Problem constants (from reference)
#define NROWS_TOTAL (4096 * 32)   // 131072 z-vectors
#define KDIM        64            // center_size
#define LCENT       1024          // number of centers
#define SIGMA       10.0f
#define SSTRIDE     1028          // padded LDS row stride (floats), 16B-aligned

// Fused soft-VQ kernel:
//   logits[r][c] = SIGMA * (2 * <z_r, cb_c> - ||cb_c||^2)   (||z||^2 cancels in softmax)
//   out[r][c]    = softmax_c(logits[r][c])
// One workgroup = 16 rows x 1024 cols. Phase 1: WMMA logits -> LDS.
// Phase 2: 16 lanes per row do the softmax and write b128-coalesced output.
__global__ __launch_bounds__(256, 1)
void svq_fused_wmma_kernel(const float* __restrict__ x,
                           const float* __restrict__ cb,
                           float* __restrict__ out) {
    __shared__ float sx[16 * KDIM];        // 16-row x tile (4 KB)
    __shared__ float ssc[16 * SSTRIDE];    // logit scores (64.3 KB)

    const int tid  = threadIdx.x;
    const int wave = tid >> 5;
    const int lane = tid & 31;
    const int n16  = lane & 15;            // row (A) / column-in-tile (B,C)
    const int h    = lane >> 4;            // lane half
    const long rowBase = (long)blockIdx.x * 16;

    // ---- stage x tile to LDS: 16 rows x 64 f32, fully contiguous ----
    {
        const float4* src = (const float4*)(x + rowBase * KDIM);
        ((float4*)sx)[tid] = src[tid];
    }
    __syncthreads();

    // ---- build A fragments (f16 hi/lo split), shared by all tiles ----
    // A k order chosen here: lane half h, elem j -> K = j + 16h
    v16h a_hi[2], a_lo[2];
#pragma unroll
    for (int kb = 0; kb < 2; ++kb) {
#pragma unroll
        for (int j = 0; j < 16; ++j) {
            int K = kb * 32 + j + 16 * h;
            float f = sx[n16 * KDIM + K];
            _Float16 hi = (_Float16)f;
            _Float16 lo = (_Float16)(f - (float)hi);
            a_hi[kb][j] = hi;
            a_lo[kb][j] = lo;
        }
    }

    // ---- Phase 1: per-tile GEMM, logits straight to LDS (no live arrays) ----
#pragma unroll 1
    for (int i = 0; i < 8; ++i) {
        const int t = wave * 8 + i;          // column tile 0..63
        const int c = t * 16 + n16;          // this lane's center index
        // B k order chosen here (matches A): lane half h, elem j -> K = j + 16h (contiguous 16 f32)
        const float4* bp4 = (const float4*)(cb + (long)c * KDIM + 16 * h);
        v16h b_hi[2], b_lo[2];
        float ss = 0.f;
#pragma unroll
        for (int kb = 0; kb < 2; ++kb) {
#pragma unroll
            for (int q = 0; q < 4; ++q) {
                float4 v = bp4[kb * 8 + q];
                float fv[4] = {v.x, v.y, v.z, v.w};
#pragma unroll
                for (int e = 0; e < 4; ++e) {
                    int j = q * 4 + e;
                    float f = fv[e];
                    _Float16 hi = (_Float16)f;
                    _Float16 lo = (_Float16)(f - (float)hi);
                    b_hi[kb][j] = hi;
                    b_lo[kb][j] = lo;
                    ss += f * f;
                }
            }
        }
        // c_sq = this half's squares + other half's (lane ^ 16)
        const float csq = ss + __shfl_xor(ss, 16, 32);

        v8f a = {};
        a = wmma_f16(a_hi[0], b_hi[0], a);
        a = wmma_f16(a_hi[1], b_hi[1], a);
        a = wmma_f16(a_lo[0], b_hi[0], a);
        a = wmma_f16(a_lo[1], b_hi[1], a);
        a = wmma_f16(a_hi[0], b_lo[0], a);
        a = wmma_f16(a_hi[1], b_lo[1], a);

        // logits -> LDS: C VGPR r holds row (r + 8h), this lane's col t*16+n16
        float* dst = &ssc[t * 16 + n16 + 8 * h * SSTRIDE];
#pragma unroll
        for (int r = 0; r < 8; ++r)
            dst[r * SSTRIDE] = (2.0f * SIGMA) * a[r] - SIGMA * csq;
    }
    __syncthreads();

    // ---- Phase 2: softmax, 16 lanes per row, 64 cols each ----
    const int row = tid >> 4;               // 0..15
    const int l16 = tid & 15;
    const float* src = &ssc[row * SSTRIDE + l16 * 4];

    float4 vals[16];
    float mx = -__builtin_inff();
#pragma unroll
    for (int q = 0; q < 16; ++q) {
        float4 v = *(const float4*)(src + q * 64);
        vals[q] = v;
        mx = fmaxf(mx, fmaxf(fmaxf(v.x, v.y), fmaxf(v.z, v.w)));
    }
#pragma unroll
    for (int msk = 1; msk <= 8; msk <<= 1)
        mx = fmaxf(mx, __shfl_xor(mx, msk, 32));

    float s = 0.f;
#pragma unroll
    for (int q = 0; q < 16; ++q) {
        float4 v = vals[q];
        v.x = __expf(v.x - mx);
        v.y = __expf(v.y - mx);
        v.z = __expf(v.z - mx);
        v.w = __expf(v.w - mx);
        vals[q] = v;
        s += (v.x + v.y) + (v.z + v.w);
    }
#pragma unroll
    for (int msk = 1; msk <= 8; msk <<= 1)
        s += __shfl_xor(s, msk, 32);
    const float inv = __frcp_rn(s);

    // ---- store: b128, 256B contiguous per half-wave ----
    float* orow = out + (rowBase + row) * (long)LCENT + l16 * 4;
#pragma unroll
    for (int q = 0; q < 16; ++q) {
        float4 v = vals[q];
        v.x *= inv; v.y *= inv; v.z *= inv; v.w *= inv;
        *(float4*)(orow + q * 64) = v;
    }
}

extern "C" void kernel_launch(void* const* d_in, const int* in_sizes, int n_in,
                              void* d_out, int out_size, void* d_ws, size_t ws_size,
                              hipStream_t stream) {
    (void)in_sizes; (void)n_in; (void)d_ws; (void)ws_size; (void)out_size;
    const float* x  = (const float*)d_in[0];     // (4096, 2048) f32
    const float* cb = (const float*)d_in[1];     // (1024, 64)  f32
    float* out = (float*)d_out;                  // (4096, 32, 1024) f32

    const int blocks = NROWS_TOTAL / 16;         // 8192 workgroups of 256 threads
    svq_fused_wmma_kernel<<<blocks, 256, 0, stream>>>(x, cb, out);
}
